// SelfAttention_11201274708744
// MI455X (gfx1250) — matrix operand order repaired and checked by reading, not compiled, not run on hardware
//
#include <hip/hip_runtime.h>

typedef __attribute__((ext_vector_type(16))) _Float16 v16h;
typedef __attribute__((ext_vector_type(8)))  _Float16 v8h;
typedef __attribute__((ext_vector_type(8)))  float    v8f;

constexpr int BATCH = 8;
constexpr int CH    = 256;   // value/channel dim
constexpr int CQ    = 32;    // head dim (C/8)
constexpr int NPOS  = 4096;  // H*W

// ---------------------------------------------------------------------------
// Projection GEMM:  out[o, n] = sum_c W[o, c] * x[b, c, n] + bias[o]
// One wave per (OT*16)(o) x 64(n) strip; each x tile (B operand) is loaded
// and converted once and reused across OT A-operand tiles.
// ---------------------------------------------------------------------------
template <int OT>
__global__ __launch_bounds__(32)
void proj_f16_kernel(const float* __restrict__ x, const float* __restrict__ w,
                     const float* __restrict__ bias, _Float16* __restrict__ out,
                     int O, int posMajor)
{
    const int lane = threadIdx.x & 31;
    const int hi   = lane >> 4;
    const int lm   = lane & 15;
    const int b    = blockIdx.z;
    const int o0   = blockIdx.y * (OT * 16);
    const int n0   = blockIdx.x * 64;

    v8f acc[OT][4] = {};
    const float* xb = x + (size_t)b * CH * NPOS;

    for (int c0 = 0; c0 < CH; c0 += 32) {
        // B operands: x tiles, 32(c) x 16(n). lane -> N = lm ; slot j -> K = 16*hi + j
        v16h bX[4];
        #pragma unroll
        for (int t = 0; t < 4; ++t) {
            int n = n0 + t * 16 + lm;
            #pragma unroll
            for (int j = 0; j < 16; ++j) {
                int c = c0 + hi * 16 + j;
                bX[t][j] = (_Float16)xb[(size_t)c * NPOS + n];
            }
        }
        #pragma unroll
        for (int ot = 0; ot < OT; ++ot) {
            // A operand: W tile, 16(o) x 32(c). lane -> M = lm ; slot j -> K = 16*hi + j
            v16h aW;
            #pragma unroll
            for (int j = 0; j < 16; ++j) {
                int c = c0 + hi * 16 + j;
                aW[j] = (_Float16)w[(o0 + ot * 16 + lm) * CH + c];
            }
            #pragma unroll
            for (int t = 0; t < 4; ++t)
                acc[ot][t] = __builtin_amdgcn_wmma_f32_16x16x32_f16(
                    false, aW, false, bX[t], (short)0, acc[ot][t], false, false);
        }
    }

    // D layout: VGPR r -> o = base + r + 8*hi ; lane -> n = n0 + 16t + lm
    #pragma unroll
    for (int ot = 0; ot < OT; ++ot) {
        #pragma unroll
        for (int r = 0; r < 8; ++r) {
            int o = o0 + ot * 16 + r + 8 * hi;
            float bv = bias[o];
            #pragma unroll
            for (int t = 0; t < 4; ++t) {
                int n = n0 + t * 16 + lm;
                float v = acc[ot][t][r] + bv;
                if (posMajor)
                    out[((size_t)b * NPOS + n) * O + o] = (_Float16)v;   // [B][N][O]
                else
                    out[((size_t)b * O + o) * NPOS + n] = (_Float16)v;   // [B][O][N]
            }
        }
    }
}

// ---------------------------------------------------------------------------
// Fused flash attention (softmax over keys n, per query m):
//   keys   fK [B][N][32], queries gQ [B][N][32], values hV [B][C][N] (f16)
//   out [B][C][N] f32
// 8 waves per WG; each wave owns 16 queries and all 256 channels.
// Online softmax, 64 keys per step: 4 score WMMAs + 32 P*V WMMAs.
// ---------------------------------------------------------------------------
__global__ __launch_bounds__(256)
void flash_attn_kernel(const _Float16* __restrict__ fK,
                       const _Float16* __restrict__ gQ,
                       const _Float16* __restrict__ hV,
                       float* __restrict__ out)
{
    const int tid  = threadIdx.x;
    const int wave = tid >> 5;
    const int lane = tid & 31;
    const int hi   = lane >> 4;
    const int lm   = lane & 15;
    const int b    = blockIdx.y;
    const int m0   = blockIdx.x * 128 + wave * 16;

    __shared__ float                sS[8][16][67];   // raw scores, odd stride (bank-safe)
    __shared__ alignas(16) _Float16 sP[8][16][72];   // exp(scores) f16
    __shared__ float                sRed[8][32];     // cross-half partial max/sum
    __shared__ float                sBcast[8][16];   // running-max / inv-sum broadcast
    __shared__ float                sScale[8][16];   // per-row accumulator rescale

    // ---- Q as WMMA A operand (16 queries x 32 dims), held for whole kernel
    v16h aQ;
    {
        const _Float16* qrow = gQ + ((size_t)b * NPOS + (m0 + lm)) * CQ;
        #pragma unroll
        for (int j = 0; j < 16; ++j) {
            int d = hi * 16 + j;
            aQ[j] = qrow[d];
        }
    }

    v8f acc[16] = {};          // 16 queries x 256 channels, f32
    const v8f zero = {};
    float rowMax = -1e30f;     // valid in lanes 0..15 (row = lm)
    float rowSum = 0.0f;

    for (int n0 = 0; n0 < NPOS; n0 += 64) {
        // ---- scores: S[m, k] for 64 keys (four 16-key WMMA tiles)
        v8f s[4];
        #pragma unroll
        for (int tt = 0; tt < 4; ++tt) {
            const v16h bK = *(const v16h*)(fK + ((size_t)b * NPOS + n0 + tt * 16 + lm) * CQ + hi * 16);
            s[tt] = __builtin_amdgcn_wmma_f32_16x16x32_f16(false, aQ, false, bK,
                                                           (short)0, zero, false, false);
        }
        if (n0 + 64 < NPOS)   // prefetch next key block
            __builtin_prefetch(fK + ((size_t)b * NPOS + n0 + 64 + lm) * CQ, 0, 1);

        // D layout -> LDS rows: row m = r + 8*hi, col = key-in-block
        #pragma unroll
        for (int tt = 0; tt < 4; ++tt)
            #pragma unroll
            for (int r = 0; r < 8; ++r)
                sS[wave][r + 8 * hi][tt * 16 + lm] = s[tt][r];
        __syncthreads();

        // ---- pass 1: per-row block max (lane handles row lm, cols hi*32..hi*32+31)
        float pm = -1e30f;
        #pragma unroll
        for (int c = 0; c < 32; ++c) pm = fmaxf(pm, sS[wave][lm][hi * 32 + c]);
        sRed[wave][lane] = pm;
        __syncthreads();

        if (lane < 16) {
            float bm = fmaxf(sRed[wave][lm], sRed[wave][16 + lm]);
            float nm = fmaxf(rowMax, bm);
            sScale[wave][lm] = __expf(rowMax - nm);  // == 1.0f exactly if max unchanged
            sBcast[wave][lm] = nm;
            rowMax = nm;
        }
        __syncthreads();

        // ---- pass 2: p = exp(s - max), partial sums, pack P to f16
        float nm = sBcast[wave][lm];
        float ps = 0.0f;
        #pragma unroll
        for (int c = 0; c < 32; ++c) {
            int col = hi * 32 + c;
            float p = __expf(sS[wave][lm][col] - nm);
            ps += p;
            sP[wave][lm][col] = (_Float16)p;
        }
        sRed[wave][lane] = ps;
        __syncthreads();

        if (lane < 16)
            rowSum = rowSum * sScale[wave][lm] + sRed[wave][lm] + sRed[wave][16 + lm];

        // ---- rescale accumulators (skipped when no row's max moved: scale==1)
        float sc[8];
        bool need = false;
        #pragma unroll
        for (int r = 0; r < 8; ++r) {
            sc[r] = sScale[wave][r + 8 * hi];
            need |= (sc[r] != 1.0f);
        }
        if (need) {
            #pragma unroll
            for (int r = 0; r < 8; ++r)
                #pragma unroll
                for (int t = 0; t < 16; ++t) acc[t][r] *= sc[r];
        }

        // ---- repack P into two 16x32 A operands (keys 0..31 and 32..63)
        //      lane m = lm; slot j -> key kb + 16*hi + j
        v8h p00 = *(const v8h*)&sP[wave][lm][hi * 16];
        v8h p01 = *(const v8h*)&sP[wave][lm][8 + hi * 16];
        v8h p10 = *(const v8h*)&sP[wave][lm][32 + hi * 16];
        v8h p11 = *(const v8h*)&sP[wave][lm][40 + hi * 16];
        v16h aP0, aP1;
        #pragma unroll
        for (int j = 0; j < 8; ++j) {
            aP0[j] = p00[j]; aP0[j + 8] = p01[j];
            aP1[j] = p10[j]; aP1[j + 8] = p11[j];
        }

        // ---- O += P * V : 16 channel tiles x 2 key chunks, V contiguous in [B][C][N]
        const _Float16* vbase = hV + (size_t)b * CH * NPOS + n0 + hi * 16;
        #pragma unroll
        for (int t = 0; t < 16; ++t) {
            const v16h bV = *(const v16h*)(vbase + (size_t)(t * 16 + lm) * NPOS);
            acc[t] = __builtin_amdgcn_wmma_f32_16x16x32_f16(false, aP0, false, bV,
                                                            (short)0, acc[t], false, false);
        }
        #pragma unroll
        for (int t = 0; t < 16; ++t) {
            const v16h bV = *(const v16h*)(vbase + (size_t)(t * 16 + lm) * NPOS + 32);
            acc[t] = __builtin_amdgcn_wmma_f32_16x16x32_f16(false, aP1, false, bV,
                                                            (short)0, acc[t], false, false);
        }
        __syncthreads();   // protect sS/sP/sScale reuse next iteration
    }

    // ---- epilogue: normalize by row sum, store out[b][c][m] f32
    if (lane < 16) sBcast[wave][lm] = 1.0f / rowSum;
    __syncthreads();

    #pragma unroll
    for (int r = 0; r < 8; ++r) {
        float inv = sBcast[wave][r + 8 * hi];
        int m = m0 + r + 8 * hi;
        #pragma unroll
        for (int t = 0; t < 16; ++t) {
            int c = t * 16 + lm;
            out[((size_t)b * CH + c) * NPOS + m] = acc[t][r] * inv;
        }
    }
}

// ---------------------------------------------------------------------------
extern "C" void kernel_launch(void* const* d_in, const int* in_sizes, int n_in,
                              void* d_out, int out_size, void* d_ws, size_t ws_size,
                              hipStream_t stream)
{
    const float* x  = (const float*)d_in[0];
    const float* wq = (const float*)d_in[1];
    const float* bq = (const float*)d_in[2];
    const float* wk = (const float*)d_in[3];
    const float* bk = (const float*)d_in[4];
    const float* wv = (const float*)d_in[5];
    const float* bv = (const float*)d_in[6];
    float* out = (float*)d_out;

    // Workspace layout (f16): keys 2MB | queries 2MB | values 16MB
    char* ws = (char*)d_ws;
    _Float16* fK = (_Float16*)(ws);                                  // [B][N][32]
    _Float16* gQ = (_Float16*)(ws + (size_t)2 * 1024 * 1024);        // [B][N][32]
    _Float16* hV = (_Float16*)(ws + (size_t)4 * 1024 * 1024);        // [B][C][N]

    // scores[n,m] = f[:,n]·g[:,m]; softmax over n  =>  keys come from wq (f),
    // queries from wk (g), values from wv (h).
    proj_f16_kernel<2><<<dim3(NPOS / 64, 1, BATCH), dim3(32), 0, stream>>>(
        x, wq, bq, fK, CQ, /*posMajor=*/1);
    proj_f16_kernel<2><<<dim3(NPOS / 64, 1, BATCH), dim3(32), 0, stream>>>(
        x, wk, bk, gQ, CQ, /*posMajor=*/1);
    proj_f16_kernel<4><<<dim3(NPOS / 64, CH / 64, BATCH), dim3(32), 0, stream>>>(
        x, wv, bv, hV, CH, /*posMajor=*/0);

    flash_attn_kernel<<<dim3(NPOS / 128, BATCH), dim3(256), 0, stream>>>(fK, gQ, hV, out);
}
